// MultiHeadSelfAttention_58402965291293
// MI455X (gfx1250) — matrix operand order repaired and checked by reading, not compiled, not run on hardware
//
#include <hip/hip_runtime.h>
#include <hip/hip_bf16.h>

typedef __attribute__((ext_vector_type(16))) __bf16       bf16x16;
typedef __attribute__((ext_vector_type(8)))  float        f32x8;
typedef __attribute__((ext_vector_type(4)))  float        f32x4;
typedef __attribute__((ext_vector_type(4)))  unsigned int u32x4;
typedef __attribute__((ext_vector_type(2)))  unsigned int u32x2;

#define D_MODEL 1024
#define NHEADS  16
#define HDIM    64
#define SEQ     2048
#define BATCH   4
#define MTOT    (BATCH * SEQ)   // 8192

// ---------------------------------------------------------------- helpers ---
__device__ __forceinline__ unsigned short f2bf_bits(float f) {  // RNE f32->bf16
  union { float f; unsigned u; } x;
  x.f = f;
  unsigned r = x.u + 0x7FFFu + ((x.u >> 16) & 1u);
  return (unsigned short)(r >> 16);
}
__device__ __forceinline__ __hip_bfloat16 toBf(float x) { return __float2bfloat16(x); }

// 4 consecutive source elements -> 4 packed bf16 (2 dwords)
__device__ __forceinline__ u32x2 load_pack4(const float* p) {
  f32x4 v = *(const f32x4*)p;
  u32x2 r;
  r.x = (unsigned)f2bf_bits(v.x) | ((unsigned)f2bf_bits(v.y) << 16);
  r.y = (unsigned)f2bf_bits(v.z) | ((unsigned)f2bf_bits(v.w) << 16);
  return r;
}
__device__ __forceinline__ u32x2 load_pack4(const __hip_bfloat16* p) {
  return *(const u32x2*)p;
}

// Build a 16-element bf16 fragment from two 16-byte chunks (LDS or global).
__device__ __forceinline__ bf16x16 frag2(const __hip_bfloat16* p0, const __hip_bfloat16* p1) {
  union { u32x4 u[2]; bf16x16 v; } x;
  x.u[0] = *(const u32x4*)p0;
  x.u[1] = *(const u32x4*)p1;
  return x.v;
}
__device__ __forceinline__ f32x8 wmma_bf16(bf16x16 a, bf16x16 b, f32x8 c) {
  return __builtin_amdgcn_wmma_f32_16x16x32_bf16(false, a, false, b, (short)0, c, false, false);
}
// A-fragment (16x32, M = lane&15): elems 0..7 -> K = kbase+8*half+e ; 8..15 -> +16
__device__ __forceinline__ bf16x16 frag_a(const __hip_bfloat16* row, int half, int kbase) {
  return frag2(row + kbase + 8 * half, row + kbase + 16 + 8 * half);
}
// B-fragment (32x16, N = lane&15): elems 0..7 -> K = kbase+8*half+e ; 8..15 -> +16
__device__ __forceinline__ bf16x16 frag_b(const __hip_bfloat16* row, int half, int kbase) {
  return frag2(row + kbase + 8 * half, row + kbase + 16 + 8 * half);
}

// ------------------------------------------------------- projection GEMM ----
// Y[m,n] = sum_k A[m,k] * W[n,k] + bias[n]   (A: [8192,1024], W,b per torch Linear)
// MODE 0: fp32 out row-major [M, D_MODEL]
// MODE 1: bf16 out at [b,h,s,d]   (Q/K heads)
// MODE 2: bf16 out at [b,h,d,s]   (V transposed per head)
template <int MODE, typename TA>
__global__ __launch_bounds__(256) void gemm_wmma(const TA* __restrict__ A,
                                                 const float* __restrict__ W,
                                                 const float* __restrict__ bias,
                                                 void* __restrict__ out) {
  __shared__ __hip_bfloat16 As[64 * 72];  // 64 rows x 64 K, stride 72 (144B = 9*16B)
  __shared__ __hip_bfloat16 Bs[64 * 72];

  const int tid = threadIdx.x;
  const int lane = tid & 31, wv = tid >> 5;
  const int half = lane >> 4, l15 = lane & 15;
  const int m0 = blockIdx.x * 64;   // 128 blocks
  const int n0 = blockIdx.y * 64;   // 16 blocks
  const int tm = wv & 3;            // A sub-tile (16 rows)
  const int tn0 = (wv >> 2) * 2;    // two B sub-tiles per wave

  f32x8 c0, c1;
  {
    float b0 = bias[n0 + tn0 * 16 + l15];
    float b1 = bias[n0 + (tn0 + 1) * 16 + l15];
    for (int r = 0; r < 8; ++r) { c0[r] = b0; c1[r] = b1; }
  }

  for (int k0 = 0; k0 < D_MODEL; k0 += 64) {
    // stage 64x64 tiles as bf16: 16 float4 groups per row, 4 groups per thread
    for (int i = tid; i < 64 * 16; i += 256) {
      int r = i >> 4, c4 = (i & 15) * 4;
      *(u32x2*)&As[r * 72 + c4] = load_pack4(A + (size_t)(m0 + r) * D_MODEL + k0 + c4);
      *(u32x2*)&Bs[r * 72 + c4] = load_pack4(W + (size_t)(n0 + r) * D_MODEL + k0 + c4);
    }
    if (k0 + 64 < D_MODEL) {  // gfx1250 global_prefetch of next K slab
      int r = tid & 63;
      __builtin_prefetch(A + (size_t)(m0 + r) * D_MODEL + k0 + 64, 0, 1);
      __builtin_prefetch(W + (size_t)(n0 + r) * D_MODEL + k0 + 64, 0, 1);
    }
    __syncthreads();

    const __hip_bfloat16* arow  = &As[(tm * 16 + l15) * 72];
    const __hip_bfloat16* brow0 = &Bs[(tn0 * 16 + l15) * 72];
    const __hip_bfloat16* brow1 = &Bs[((tn0 + 1) * 16 + l15) * 72];
    bf16x16 a0 = frag_a(arow, half, 0),  a1 = frag_a(arow, half, 32);
    bf16x16 b00 = frag_b(brow0, half, 0), b01 = frag_b(brow0, half, 32);
    bf16x16 b10 = frag_b(brow1, half, 0), b11 = frag_b(brow1, half, 32);
    c0 = wmma_bf16(a0, b00, c0);
    c0 = wmma_bf16(a1, b01, c0);
    c1 = wmma_bf16(a0, b10, c1);
    c1 = wmma_bf16(a1, b11, c1);
    __syncthreads();
  }

  float* outF = (float*)out;
  __hip_bfloat16* outB = (__hip_bfloat16*)out;
  for (int t = 0; t < 2; ++t) {
    f32x8 cc = t ? c1 : c0;
    int gn = n0 + (tn0 + t) * 16 + l15;
    for (int r = 0; r < 8; ++r) {
      int gm = m0 + tm * 16 + r + 8 * half;
      float v = cc[r];
      if (MODE == 0) {
        outF[(size_t)gm * D_MODEL + gn] = v;
      } else {
        int b = gm >> 11, s = gm & (SEQ - 1);
        int h = gn >> 6,  d = gn & (HDIM - 1);
        size_t idx = (MODE == 1)
            ? (((size_t)(b * NHEADS + h) * SEQ + s) * HDIM + d)      // [b,h,s,d]
            : (((size_t)(b * NHEADS + h) * HDIM + d) * SEQ + s);     // [b,h,d,s]
        outB[idx] = toBf(v);
      }
    }
  }
}

// ----------------------------------------------------- fused attention ------
// One workgroup = one (batch, head, 16-row q tile). 8 waves.
// LDS: scores fp32 [16][2048] (128K) | P bf16 [16][2048] (64K) |
//      q bf16 [16][64] (2K) | ctx partials fp32 [8][16][64] (32K) = 226K (CDNA5 320K WGP LDS).
#define SMEM_SCORES 0
#define SMEM_PBF    (16 * SEQ * 4)
#define SMEM_QLDS   (SMEM_PBF + 16 * SEQ * 2)
#define SMEM_CTX    (SMEM_QLDS + 16 * HDIM * 2)
#define SMEM_TOTAL  (SMEM_CTX + 8 * 16 * HDIM * 4)

__global__ __launch_bounds__(256) void attn_fused(const __hip_bfloat16* __restrict__ q,
                                                  const __hip_bfloat16* __restrict__ k,
                                                  const __hip_bfloat16* __restrict__ vT,
                                                  float* __restrict__ attn_out,
                                                  __hip_bfloat16* __restrict__ ctx_out) {
  extern __shared__ char smem[];
  float*          scores = (float*)(smem + SMEM_SCORES);
  __hip_bfloat16* p_bf   = (__hip_bfloat16*)(smem + SMEM_PBF);
  __hip_bfloat16* q_lds  = (__hip_bfloat16*)(smem + SMEM_QLDS);
  float*          ctxp   = (float*)(smem + SMEM_CTX);

  const int tid = threadIdx.x;
  const int lane = tid & 31, wv = tid >> 5;
  const int half = lane >> 4, l15 = lane & 15;
  const int bh = blockIdx.x >> 7;        // b*16 + h
  const int qt = blockIdx.x & 127;       // q tile (16 rows)
  const __hip_bfloat16* Qh = q  + (size_t)bh * SEQ * HDIM;
  const __hip_bfloat16* Kh = k  + (size_t)bh * SEQ * HDIM;
  const __hip_bfloat16* Vt = vT + (size_t)bh * HDIM * SEQ;

  // stage q tile (16 x 64 bf16 = 512 dwords, contiguous)
  for (int i = tid; i < 16 * HDIM / 2; i += 256)
    ((unsigned*)q_lds)[i] = ((const unsigned*)(Qh + (size_t)qt * 16 * HDIM))[i];
  __syncthreads();

  bf16x16 qa0 = frag_a(q_lds + l15 * HDIM, half, 0);
  bf16x16 qa1 = frag_a(q_lds + l15 * HDIM, half, 32);

  // ---- scores = (Q K^T) / sqrt(64), waves striped over k-tiles ----
  for (int kt = wv; kt < SEQ / 16; kt += 8) {
    const __hip_bfloat16* kr = Kh + (size_t)(kt * 16 + l15) * HDIM;  // B: N = lane&15
    bf16x16 b0 = frag_b(kr, half, 0);
    bf16x16 b1 = frag_b(kr, half, 32);
    f32x8 c = {};
    c = wmma_bf16(qa0, b0, c);
    c = wmma_bf16(qa1, b1, c);
    int col = kt * 16 + l15;
    for (int r = 0; r < 8; ++r) scores[(r + 8 * half) * SEQ + col] = c[r] * 0.125f;
  }
  __syncthreads();

  // ---- softmax (wave w owns rows 2w, 2w+1); vectorized float4 ----
  for (int rr = 0; rr < 2; ++rr) {
    int row = wv * 2 + rr;
    f32x4* srow4 = (f32x4*)(scores + row * SEQ);
    float mx = -3.0e38f;
    for (int c = lane; c < SEQ / 4; c += 32) {
      f32x4 v = srow4[c];
      mx = fmaxf(mx, fmaxf(fmaxf(v.x, v.y), fmaxf(v.z, v.w)));
    }
    for (int off = 16; off; off >>= 1) mx = fmaxf(mx, __shfl_xor(mx, off, 32));
    float sum = 0.f;
    for (int c = lane; c < SEQ / 4; c += 32) {
      f32x4 v = srow4[c];
      v.x = __expf(v.x - mx);
      v.y = __expf(v.y - mx);
      v.z = __expf(v.z - mx);
      v.w = __expf(v.w - mx);
      srow4[c] = v;
      sum += (v.x + v.y) + (v.z + v.w);
    }
    for (int off = 16; off; off >>= 1) sum += __shfl_xor(sum, off, 32);
    float inv = 1.0f / sum;
    f32x4* arow4 = (f32x4*)(attn_out + ((size_t)bh * SEQ + qt * 16 + row) * SEQ);
    u32x2* prow  = (u32x2*)(p_bf + row * SEQ);
    for (int c = lane; c < SEQ / 4; c += 32) {
      f32x4 v = srow4[c] * inv;
      arow4[c] = v;                         // 1 GiB attn writeback, b128 stores
      u32x2 pk;
      pk.x = (unsigned)f2bf_bits(v.x) | ((unsigned)f2bf_bits(v.y) << 16);
      pk.y = (unsigned)f2bf_bits(v.z) | ((unsigned)f2bf_bits(v.w) << 16);
      prow[c] = pk;
    }
  }
  __syncthreads();

  // ---- ctx = P @ V : waves stripe K in 32-chunks, 4 n-tiles (hd=64) ----
  f32x8 acc[4] = {};
  for (int ch = wv; ch < SEQ / 32; ch += 8) {
    int base = ch * 32;
    bf16x16 a = frag_a(p_bf + l15 * SEQ, half, base);
    for (int nt = 0; nt < 4; ++nt) {
      bf16x16 b = frag_b(Vt + (size_t)(nt * 16 + l15) * SEQ, half, base);
      acc[nt] = wmma_bf16(a, b, acc[nt]);
    }
  }
  for (int nt = 0; nt < 4; ++nt)
    for (int r = 0; r < 8; ++r)
      ctxp[(wv * 16 + r + 8 * half) * HDIM + nt * 16 + l15] = acc[nt][r];
  __syncthreads();

  // reduce 8 wave partials; store ctx as [b, s, h*64+d] bf16 for out-projection
  const int b = bh >> 4, h = bh & 15;
  for (int i = tid; i < 16 * HDIM; i += 256) {
    int row = i >> 6, d = i & 63;
    float s = 0.f;
    for (int w2 = 0; w2 < 8; ++w2) s += ctxp[(w2 * 16 + row) * HDIM + d];
    size_t m = (size_t)b * SEQ + qt * 16 + row;
    ctx_out[m * D_MODEL + h * HDIM + d] = toBf(s);
  }
}

// ----------------------------------------------------------------- host -----
extern "C" void kernel_launch(void* const* d_in, const int* in_sizes, int n_in,
                              void* d_out, int out_size, void* d_ws, size_t ws_size,
                              hipStream_t stream) {
  (void)in_sizes; (void)n_in; (void)out_size; (void)ws_size;
  const float* Q    = (const float*)d_in[0];
  const float* K    = (const float*)d_in[1];
  const float* V    = (const float*)d_in[2];
  const float* WQ_w = (const float*)d_in[3];
  const float* WQ_b = (const float*)d_in[4];
  const float* WK_w = (const float*)d_in[5];
  const float* WK_b = (const float*)d_in[6];
  const float* WV_w = (const float*)d_in[7];
  const float* WV_b = (const float*)d_in[8];
  const float* WO_w = (const float*)d_in[9];
  const float* WO_b = (const float*)d_in[10];

  const size_t headElems = (size_t)BATCH * NHEADS * SEQ * HDIM;  // 8,388,608
  __hip_bfloat16* qw = (__hip_bfloat16*)d_ws;
  __hip_bfloat16* kw = qw + headElems;
  __hip_bfloat16* vw = kw + headElems;   // stored transposed [b,h,d,s]
  __hip_bfloat16* cw = vw + headElems;   // ctx [b,s,D]

  float* out  = (float*)d_out;                       // [B,S,D]
  float* attn = out + (size_t)MTOT * D_MODEL;        // [B,H,S,S]

  dim3 gg(MTOT / 64, D_MODEL / 64), bb(256);
  gemm_wmma<1, float><<<gg, bb, 0, stream>>>(Q, WQ_w, WQ_b, qw);
  gemm_wmma<1, float><<<gg, bb, 0, stream>>>(K, WK_w, WK_b, kw);
  gemm_wmma<2, float><<<gg, bb, 0, stream>>>(V, WV_w, WV_b, vw);

  static_assert(SMEM_TOTAL <= 320 * 1024, "LDS budget");
  hipFuncSetAttribute(reinterpret_cast<const void*>(attn_fused),
                      hipFuncAttributeMaxDynamicSharedMemorySize, SMEM_TOTAL);
  attn_fused<<<dim3(BATCH * NHEADS * (SEQ / 16)), bb, SMEM_TOTAL, stream>>>(qw, kw, vw, attn, cw);

  gemm_wmma<0, __hip_bfloat16><<<gg, bb, 0, stream>>>(cw, WO_w, WO_b, out);
}
